// ResidualBlock_1864015806629
// MI455X (gfx1250) — compile-verified, matrix operand order checked by reading, not run on hardware
//
#include <hip/hip_runtime.h>
#include <math.h>

#define N_NODES  50000
#define DIM      64
#define N_EDGES  800000
#define FEAT_E   32
#define N_GRAPHS 512
#define LN_EPS   1e-5f

typedef __attribute__((ext_vector_type(16))) _Float16 v16h;
typedef __attribute__((ext_vector_type(8)))  float    v8f;

// K-pair base index for 16-bit A/B fragments (16x32 tile):
// VGPR j<4 : K = j*2 + half*8 ; VGPR j>=4 : K = 16 + (j-4)*2 + half*8
__device__ __forceinline__ int kb_pair(int j, int half) {
  return (j < 4) ? (j * 2 + half * 8) : (16 + (j - 4) * 2 + half * 8);
}

// Fast SiLU: v * sigmoid(v), using v_exp_f32 + v_rcp_f32 (no IEEE div chain).
__device__ __forceinline__ float silu(float v) {
  return v * __builtin_amdgcn_rcpf(1.0f + __expf(-v));
}

__device__ __forceinline__ v8f wmma_f16(v16h a, v16h b, v8f c) {
  return __builtin_amdgcn_wmma_f32_16x16x32_f16(false, a, false, b,
                                                (short)0, c, false, false);
}

// Build A fragment (rows row0..row0+15, K slice [k0,k0+32)) from row-major f32.
__device__ __forceinline__ v16h build_a(const float* __restrict__ base,
                                        int row_stride, int row0, int k0,
                                        int lane) {
  const int half = (lane >> 4) & 1;
  const int m = lane & 15;
  const float* rp = base + (unsigned)(row0 + m) * (unsigned)row_stride + k0;
  v16h a;
#pragma unroll
  for (int j = 0; j < 8; ++j) {
    const int kb = kb_pair(j, half);
    a[2 * j]     = (_Float16)rp[kb];
    a[2 * j + 1] = (_Float16)rp[kb + 1];
  }
  return a;
}

// Build B fragment from row-major f32 weight W (K x n_stride), column tile col0.
__device__ __forceinline__ v16h build_b(const float* __restrict__ W,
                                        int n_stride, int k0, int col0,
                                        int lane) {
  const int half = (lane >> 4) & 1;
  const int n = lane & 15;
  v16h b;
#pragma unroll
  for (int j = 0; j < 8; ++j) {
    const int kb = kb_pair(j, half);
    b[2 * j]     = (_Float16)W[(unsigned)(k0 + kb) * n_stride + col0 + n];
    b[2 * j + 1] = (_Float16)W[(unsigned)(k0 + kb + 1) * n_stride + col0 + n];
  }
  return b;
}

__global__ void zero_kernel(float* __restrict__ p, int n) {
  int i = blockIdx.x * blockDim.x + threadIdx.x;
  if (i < n) p[i] = 0.0f;
}

// Fused: e = SiLU(edge_attr @ We + be); msg = relu(x[src] + e); agg[dst] += msg
__global__ __launch_bounds__(256) void edge_conv_kernel(
    const float* __restrict__ x,          // (N, 64)
    const int*   __restrict__ src,        // (E)
    const int*   __restrict__ dst,        // (E)
    const float* __restrict__ edge_attr,  // (E, 32)
    const float* __restrict__ We,         // (32, 64)
    const float* __restrict__ be,         // (64)
    float*       __restrict__ agg)        // (N, 64) pre-zeroed
{
  const int lane   = threadIdx.x & 31;
  const int wave   = blockIdx.x * (blockDim.x >> 5) + (threadIdx.x >> 5);
  const int nWaves = gridDim.x * (blockDim.x >> 5);
  const int half   = (lane >> 4) & 1;
  const int n15    = lane & 15;

  // Persistent weight fragments: K=32 in one WMMA, 4 column tiles of 16.
  v16h bw[4];
  float bias[4];
#pragma unroll
  for (int nt = 0; nt < 4; ++nt) {
    bw[nt]   = build_b(We, DIM, 0, nt * 16, lane);
    bias[nt] = be[nt * 16 + n15];
  }

  const int nTiles = N_EDGES / 16;
  for (int tile = wave; tile < nTiles; tile += nWaves) {
    const int e0 = tile * 16;
    const v16h a = build_a(edge_attr, FEAT_E, e0, 0, lane);

    // 32-bit element offsets for gather/scatter (fit in u32 byte offsets).
    unsigned sOff[8], dOff[8];
#pragma unroll
    for (int r = 0; r < 8; ++r) {
      const int em = e0 + half * 8 + r;  // C row = r + 8*half
      sOff[r] = (unsigned)src[em] * (unsigned)DIM;
      dOff[r] = (unsigned)dst[em] * (unsigned)DIM;
    }

#pragma unroll
    for (int nt = 0; nt < 4; ++nt) {
      v8f c = {};
      c = wmma_f16(a, bw[nt], c);
      const unsigned col = nt * 16 + n15;

      // Batch the 8 gather loads so they are all in flight before any use.
      float xv[8];
#pragma unroll
      for (int r = 0; r < 8; ++r) xv[r] = x[sOff[r] + col];

      // Independent SiLU chains pipeline through the TRANS unit.
      float ev[8];
#pragma unroll
      for (int r = 0; r < 8; ++r) ev[r] = silu(c[r] + bias[nt]);

#pragma unroll
      for (int r = 0; r < 8; ++r) {
        const float msg = fmaxf(xv[r] + ev[r], 0.0f);
        atomicAdd(&agg[dOff[r] + col], msg);
      }
    }
  }
}

// Fused GIN node MLP: h = agg + (1+eps)*x; h = SiLU(h@W1+b1); h = SiLU(h@W2+b2)
// Optional epilogue (identity != null): out = relu((h + identity) * 0.5)
__global__ __launch_bounds__(256) void gin_mlp_kernel(
    const float* __restrict__ agg,       // (N, 64)
    const float* __restrict__ xin,       // (N, 64)
    const float* __restrict__ eps,       // (1)
    const float* __restrict__ W1, const float* __restrict__ b1,
    const float* __restrict__ W2, const float* __restrict__ b2,
    const float* __restrict__ identity,  // (N, 64) or nullptr
    float*       __restrict__ out)       // (N, 64)
{
  __shared__ float tile[8][16 * DIM];   // per-wave 16x64 f32 intermediate

  const int lane   = threadIdx.x & 31;
  const int wslot  = threadIdx.x >> 5;
  const int wave   = blockIdx.x * (blockDim.x >> 5) + wslot;
  const int nWaves = gridDim.x * (blockDim.x >> 5);
  const int half   = (lane >> 4) & 1;
  const int n15    = lane & 15;
  const float epsv = 1.0f + eps[0];

  // Persistent weight fragments: [kt 0..1][nt 0..3] for both layers.
  v16h w1f[2][4], w2f[2][4];
#pragma unroll
  for (int kt = 0; kt < 2; ++kt)
#pragma unroll
    for (int nt = 0; nt < 4; ++nt) {
      w1f[kt][nt] = build_b(W1, DIM, kt * 32, nt * 16, lane);
      w2f[kt][nt] = build_b(W2, DIM, kt * 32, nt * 16, lane);
    }
  float bias1[4], bias2[4];
#pragma unroll
  for (int nt = 0; nt < 4; ++nt) {
    bias1[nt] = b1[nt * 16 + n15];
    bias2[nt] = b2[nt * 16 + n15];
  }

  float* myTile = tile[wslot];
  const int nTiles = N_NODES / 16;

  for (int t = wave; t < nTiles; t += nWaves) {
    const int r0 = t * 16;

    // A = agg + (1+eps)*x for this 16-row tile; two K-chunk fragments.
    v16h a0, a1;
    {
      const unsigned ro = (unsigned)(r0 + n15) * (unsigned)DIM;
      const float* ap = agg + ro;
      const float* xp = xin + ro;
#pragma unroll
      for (int j = 0; j < 8; ++j) {
        const int kb = kb_pair(j, half);
        a0[2 * j]     = (_Float16)(ap[kb]          + epsv * xp[kb]);
        a0[2 * j + 1] = (_Float16)(ap[kb + 1]      + epsv * xp[kb + 1]);
        a1[2 * j]     = (_Float16)(ap[32 + kb]     + epsv * xp[32 + kb]);
        a1[2 * j + 1] = (_Float16)(ap[32 + kb + 1] + epsv * xp[32 + kb + 1]);
      }
    }

    // Layer 1: C = A @ W1 (K=64 via two WMMAs), SiLU, stash in LDS tile.
#pragma unroll
    for (int nt = 0; nt < 4; ++nt) {
      v8f c = {};
      c = wmma_f16(a0, w1f[0][nt], c);
      c = wmma_f16(a1, w1f[1][nt], c);
      const int col = nt * 16 + n15;
      float ev[8];
#pragma unroll
      for (int r = 0; r < 8; ++r) ev[r] = silu(c[r] + bias1[nt]);
#pragma unroll
      for (int r = 0; r < 8; ++r)
        myTile[(r + 8 * half) * DIM + col] = ev[r];
    }

    // Same-wave LDS RAW: DS ops are in-order per wave; fence the counter.
    asm volatile("s_wait_dscnt 0" ::: "memory");

    // Rebuild A fragments from the LDS tile for layer 2.
    v16h t0, t1;
    {
      const float* rp = myTile + n15 * DIM;
#pragma unroll
      for (int j = 0; j < 8; ++j) {
        const int kb = kb_pair(j, half);
        t0[2 * j]     = (_Float16)rp[kb];
        t0[2 * j + 1] = (_Float16)rp[kb + 1];
        t1[2 * j]     = (_Float16)rp[32 + kb];
        t1[2 * j + 1] = (_Float16)rp[32 + kb + 1];
      }
    }

    // Layer 2 + epilogue.
#pragma unroll
    for (int nt = 0; nt < 4; ++nt) {
      v8f c = {};
      c = wmma_f16(t0, w2f[0][nt], c);
      c = wmma_f16(t1, w2f[1][nt], c);
      const unsigned col = nt * 16 + n15;
      float ev[8];
#pragma unroll
      for (int r = 0; r < 8; ++r) ev[r] = silu(c[r] + bias2[nt]);
#pragma unroll
      for (int r = 0; r < 8; ++r) {
        const unsigned row = (unsigned)(r0 + r + 8 * half);
        const unsigned off = row * (unsigned)DIM + col;
        float v = ev[r];
        if (identity != nullptr)
          v = fmaxf((v + identity[off]) * 0.5f, 0.0f);
        out[off] = v;
      }
    }
  }
}

// Per-graph sum / sumsq / count accumulation (one thread per node).
__global__ void stats_kernel(const float* __restrict__ h,
                             const int*   __restrict__ n2g,
                             float*       __restrict__ stats) {
  const int node = blockIdx.x * blockDim.x + threadIdx.x;
  if (node >= N_NODES) return;
  const float* rp = h + (unsigned)node * (unsigned)DIM;
  float s = 0.0f, ss = 0.0f;
#pragma unroll
  for (int c = 0; c < DIM; ++c) {
    const float v = rp[c];
    s += v;
    ss += v * v;
  }
  const int g = n2g[node];
  atomicAdd(&stats[g], s);
  atomicAdd(&stats[N_GRAPHS + g], ss);
  atomicAdd(&stats[2 * N_GRAPHS + g], 1.0f);
}

__global__ void finalize_stats_kernel(float* __restrict__ stats) {
  const int g = blockIdx.x * blockDim.x + threadIdx.x;
  if (g >= N_GRAPHS) return;
  const float cnt  = stats[2 * N_GRAPHS + g];
  const float norm = fmaxf(cnt * (float)DIM, 1.0f);
  const float mean = stats[g] / norm;
  float var = stats[N_GRAPHS + g] / norm - mean * mean;
  var = fmaxf(var, 0.0f);
  stats[3 * N_GRAPHS + g] = mean;
  stats[4 * N_GRAPHS + g] = rsqrtf(var + LN_EPS);
}

// out = relu((h - mean[g]) * inv[g] * w[c] + b[c])  (in-place safe)
__global__ void apply_norm_kernel(const float* __restrict__ h,
                                  const int*   __restrict__ n2g,
                                  const float* __restrict__ stats,
                                  const float* __restrict__ w,
                                  const float* __restrict__ b,
                                  float*       __restrict__ out) {
  const int i = blockIdx.x * blockDim.x + threadIdx.x;
  if (i >= N_NODES * DIM) return;
  const int node = i >> 6;
  const int c = i & 63;
  const int g = n2g[node];
  const float mean = stats[3 * N_GRAPHS + g];
  const float inv  = stats[4 * N_GRAPHS + g];
  const float v = (h[i] - mean) * inv * w[c] + b[c];
  out[i] = fmaxf(v, 0.0f);
}

extern "C" void kernel_launch(void* const* d_in, const int* in_sizes, int n_in,
                              void* d_out, int out_size, void* d_ws,
                              size_t ws_size, hipStream_t stream) {
  (void)in_sizes; (void)n_in; (void)out_size; (void)ws_size;

  const float* x         = (const float*)d_in[0];
  const int*   ei        = (const int*)d_in[1];
  const int*   src       = ei;
  const int*   dst       = ei + N_EDGES;
  const float* edge_attr = (const float*)d_in[2];
  const int*   n2g       = (const int*)d_in[3];
  const float* We1 = (const float*)d_in[4];
  const float* be1 = (const float*)d_in[5];
  const float* ep1 = (const float*)d_in[6];
  const float* W11 = (const float*)d_in[7];
  const float* b11 = (const float*)d_in[8];
  const float* W12 = (const float*)d_in[9];
  const float* b12 = (const float*)d_in[10];
  const float* gnw = (const float*)d_in[11];
  const float* gnb = (const float*)d_in[12];
  const float* We2 = (const float*)d_in[13];
  const float* be2 = (const float*)d_in[14];
  const float* ep2 = (const float*)d_in[15];
  const float* W21 = (const float*)d_in[16];
  const float* b21 = (const float*)d_in[17];
  const float* W22 = (const float*)d_in[18];
  const float* b22 = (const float*)d_in[19];
  float* out = (float*)d_out;

  float* ws    = (float*)d_ws;
  float* agg   = ws;                              // N*D
  float* h1    = agg + (size_t)N_NODES * DIM;     // N*D (also reused as x2)
  float* stats = h1 + (size_t)N_NODES * DIM;      // 5*G

  const int ND = N_NODES * DIM;

  // ---- conv1 ----
  zero_kernel<<<(ND + 255) / 256, 256, 0, stream>>>(agg, ND);
  zero_kernel<<<(5 * N_GRAPHS + 255) / 256, 256, 0, stream>>>(stats,
                                                              5 * N_GRAPHS);
  edge_conv_kernel<<<1250, 256, 0, stream>>>(x, src, dst, edge_attr, We1, be1,
                                             agg);
  gin_mlp_kernel<<<391, 256, 0, stream>>>(agg, x, ep1, W11, b11, W12, b12,
                                          nullptr, h1);
  // ---- graph norm + relu (in place on h1) ----
  stats_kernel<<<(N_NODES + 255) / 256, 256, 0, stream>>>(h1, n2g, stats);
  finalize_stats_kernel<<<2, 256, 0, stream>>>(stats);
  apply_norm_kernel<<<(ND + 255) / 256, 256, 0, stream>>>(h1, n2g, stats, gnw,
                                                          gnb, h1);
  // ---- conv2 (input = h1) + residual epilogue ----
  zero_kernel<<<(ND + 255) / 256, 256, 0, stream>>>(agg, ND);
  edge_conv_kernel<<<1250, 256, 0, stream>>>(h1, src, dst, edge_attr, We2, be2,
                                             agg);
  gin_mlp_kernel<<<391, 256, 0, stream>>>(agg, h1, ep2, W21, b21, W22, b22,
                                          x /*identity*/, out);
}
